// PAFM_16862041604769
// MI455X (gfx1250) — compile-verified, matrix operand order checked by reading, not run on hardware
//
#include <hip/hip_runtime.h>

typedef __attribute__((ext_vector_type(16))) _Float16 v16h;
typedef __attribute__((ext_vector_type(8)))  _Float16 v8h;
typedef __attribute__((ext_vector_type(8)))  float    v8f;

#define BDIM 4
#define DIM  128
#define CH   256
#define HH_  256
#define WW_  256
#define H0_  128
#define W0_  128
#define N0_  (H0_ * W0_)      /* 16384 */
#define NH_  8
#define HD_  16
#define PATCHES 32            /* patches per block -> 128 src rows, 32 tgt rows */

/* workspace layout (halves for f16 weights, then fp32 scratch) */
#define WSH_WQ   0            /* 128x128 */
#define WSH_WQV  16384        /* 128x128 */
#define WSH_WK   32768        /* 128x256 */
#define WSH_WV   65536        /* 128x256 */
#define WSH_WP   98304        /* 128x128 */
#define WSH_TOT  114688
#define WSB_MEAN 229376       /* bytes: 2*4*256 f32 */
#define WSB_CM   237568       /* bytes: 2*4*8*16 f32 */

/* dynamic LDS partition (bytes) -- 128KB total => 2 workgroups per 320KB WGP */
#define SM_SRC   0            /* f16 128x128 = 32768 ; aliased as sX later  */
#define SM_TGT   32768        /* f16  32x256 = 16384                         */
#define SM_K     49152        /* f16  32x128 =  8192                         */
#define SM_V     57344        /* f16  32x128 =  8192                         */
#define SM_Q     65536        /* f16 128x128 = 32768 ; sP (f32) aliases here */
#define SM_VQ    98304        /* f16 128x128 = 32768 ; ...and here           */
#define SM_P     65536        /* f32 128x128 = 65536 (alias over sQ+sVQ)     */
#define SM_TOTAL 131072

/* ---- WMMA helpers ------------------------------------------------------- */

__device__ inline v16h frag_ld(const _Float16* base, int ld, int rc, int kb, int lane) {
  // 16x32 f16 fragment (A with rc=row-tile base, or B=W^T with rc=col/W-row base)
  // lane l<16:  M/N = rc+l,    halves K = kb+0..7  and kb+16..23
  // lane l>=16: M/N = rc+l-16, halves K = kb+8..15 and kb+24..31
  int l16 = lane & 15;
  int hi  = (lane >> 4) & 1;
  const _Float16* p = base + (size_t)(rc + l16) * ld + kb + hi * 8;
  union { v16h v; v8h h[2]; } u;
  u.h[0] = *(const v8h*)(p);
  u.h[1] = *(const v8h*)(p + 16);
  return u.v;
}

__device__ inline v8f wmma_f16(v16h a, v16h b, v8f c) {
  return __builtin_amdgcn_wmma_f32_16x16x32_f16(false, a, false, b, (short)0, c,
                                                false, false);
}

__device__ inline void store_tile_f16(_Float16* dst, int ld, int tm, int tn, v8f acc,
                                      int lane) {
  int l16 = lane & 15;
  int hi  = (lane >> 4) & 1;
#pragma unroll
  for (int j = 0; j < 8; ++j)
    dst[(size_t)(tm + hi * 8 + j) * ld + tn + l16] = (_Float16)acc[j];
}

__device__ inline void store_tile_f32(float* dst, int ld, int tm, int tn, v8f acc,
                                      int lane) {
  int l16 = lane & 15;
  int hi  = (lane >> 4) & 1;
#pragma unroll
  for (int j = 0; j < 8; ++j)
    dst[(size_t)(tm + hi * 8 + j) * ld + tn + l16] = acc[j];
}

/* 16 contiguous halves <-> fp32 registers (two b128 LDS ops) */
__device__ inline void ld16h(const _Float16* p, float* dst) {
  v8h a = *(const v8h*)p;
  v8h b = *(const v8h*)(p + 8);
#pragma unroll
  for (int i = 0; i < 8; ++i) { dst[i] = (float)a[i]; dst[8 + i] = (float)b[i]; }
}

__device__ inline void st16h(_Float16* p, const float* src) {
  v8h a, b;
#pragma unroll
  for (int i = 0; i < 8; ++i) {
    a[i] = (_Float16)src[i];
    b[i] = (_Float16)src[8 + i];
  }
  *(v8h*)p = a;
  *(v8h*)(p + 8) = b;
}

/* ---- kernel 1: fp32 -> f16 weight conversion ---------------------------- */

__global__ void k_convert(const float* wq, const float* wqv, const float* wk,
                          const float* wv, const float* wp, _Float16* dst) {
  int i = blockIdx.x * 256 + threadIdx.x;
  if (i < 16384)        dst[i] = (_Float16)wq[i];
  else if (i < 32768)   dst[i] = (_Float16)wqv[i - 16384];
  else if (i < 65536)   dst[i] = (_Float16)wk[i - 32768];
  else if (i < 98304)   dst[i] = (_Float16)wv[i - 65536];
  else if (i < WSH_TOT) dst[i] = (_Float16)wp[i - 98304];
}

/* ---- kernel 2: channel-mean of target over N0 --------------------------- */

__global__ void k_reduce_tgt(const float* xl, const float* xr, float* mean_out) {
  int blk  = blockIdx.x;          // 0..7 = side*4 + b
  int side = blk >> 2;
  int b    = blk & 3;
  int c    = threadIdx.x;         // 0..255
  const float* x = side ? xr : xl;
  const float* p = x + ((size_t)b * CH + c) * N0_;
  float s = 0.f;
  for (int n = 0; n < N0_; ++n) s += p[n];
  mean_out[(size_t)blk * CH + c] = s * (1.0f / N0_);
}

/* ---- kernel 3: DANE channel gate (km -> silu -> LN -> proj) ------------- */

__global__ void k_cm(const float* mean_tgt, const float* Wk, const float* bk,
                     const float* w1, const float* ln2g, const float* ln2b,
                     const float* w2, float* cm) {
  int t = threadIdx.x;            // 0..63 = (side, b, head)
  if (t >= 64) return;
  int side = t >> 5;
  int rem  = t & 31;
  int b    = rem >> 3;
  int hh   = rem & 7;
  const float* mt = mean_tgt + (size_t)((side << 2) + b) * CH;
  float km[HD_];
#pragma unroll
  for (int d = 0; d < HD_; ++d) {
    int nrow = hh * HD_ + d;
    float s = bk[nrow];
    const float* wr = Wk + (size_t)nrow * CH;
    for (int c = 0; c < CH; ++c) s += mt[c] * wr[c];
    km[d] = s;
  }
  float t1[8];
#pragma unroll
  for (int r = 0; r < 8; ++r) {
    float s = 0.f;
    for (int d = 0; d < HD_; ++d) s += km[d] * w1[r * HD_ + d];
    t1[r] = s / (1.f + __expf(-s));          // silu
  }
  float mu = 0.f;
  for (int r = 0; r < 8; ++r) mu += t1[r];
  mu *= 0.125f;
  float var = 0.f;
  for (int r = 0; r < 8; ++r) { float dd = t1[r] - mu; var += dd * dd; }
  var *= 0.125f;
  float inv = rsqrtf(var + 1e-5f);
#pragma unroll
  for (int d = 0; d < HD_; ++d) {
    float s = 0.f;
    for (int r = 0; r < 8; ++r)
      s += ((t1[r] - mu) * inv * ln2g[r] + ln2b[r]) * w2[d * 8 + r];
    cm[t * HD_ + d] = s;
  }
}

/* ---- kernel 4: fused cross-attention main kernel ------------------------ */

__global__ __launch_bounds__(256)
void k_pafm_main(const float* xl, const float* xr,
                 const float* xlh, const float* xrh,
                 const _Float16* wf16,
                 const float* bq, const float* bqv, const float* bkk,
                 const float* bv, const float* bp,
                 const float* lng, const float* lnb,
                 const float* ln1g, const float* ln1b, const float* dwsw,
                 const float* cm, float* out) {
  extern __shared__ char smem[];
  _Float16* sSrc = (_Float16*)(smem + SM_SRC);
  _Float16* sTgt = (_Float16*)(smem + SM_TGT);
  _Float16* sK   = (_Float16*)(smem + SM_K);
  _Float16* sV   = (_Float16*)(smem + SM_V);
  _Float16* sQ   = (_Float16*)(smem + SM_Q);
  _Float16* sVQ  = (_Float16*)(smem + SM_VQ);
  _Float16* sX   = sSrc;                    // alias (src dead after phase 1)
  float*    sP   = (float*)(smem + SM_P);   // alias (Q/VQ dead after phase 3)

  const _Float16* Wq_h  = wf16 + WSH_WQ;
  const _Float16* Wqv_h = wf16 + WSH_WQV;
  const _Float16* Wk_h  = wf16 + WSH_WK;
  const _Float16* Wv_h  = wf16 + WSH_WV;
  const _Float16* Wp_h  = wf16 + WSH_WP;

  int tid  = threadIdx.x;
  int lane = tid & 31;
  int wv   = tid >> 5;

  const int blocksPerB = N0_ / PATCHES;     // 512
  int blk  = blockIdx.x;
  int side = blk / (BDIM * blocksPerB);
  int rem  = blk % (BDIM * blocksPerB);
  int b    = rem / blocksPerB;
  int p0   = (rem % blocksPerB) * PATCHES;

  const float* xh  = side ? xrh : xlh;
  const float* xlo = side ? xr  : xl;

  /* -- phase 0: stage src (2x2 patchified high-res) and tgt as f16 -- */
  for (int idx = tid; idx < 128 * DIM; idx += 256) {
    int r = idx & 127, c = idx >> 7;
    int p = r >> 2, s = r & 3;
    int n = p0 + p;
    int ph = n >> 7, pw = n & 127;
    int yy = ph * 2 + (s >> 1), xx = pw * 2 + (s & 1);
    sSrc[r * DIM + c] =
        (_Float16)xh[(((size_t)b * DIM + c) * HH_ + yy) * WW_ + xx];
  }
  for (int idx = tid; idx < PATCHES * CH; idx += 256) {
    int r = idx & 31, c = idx >> 5;
    sTgt[r * CH + c] =
        (_Float16)xlo[((size_t)b * CH + c) * N0_ + (p0 + r)];
  }
  __syncthreads();

  /* -- phase 1: Q and VQ GEMMs (128x128 each), waves 0-3 Q, 4-7 VQ -- */
  for (int i = 0; i < 16; ++i) {
    int t   = wv * 16 + i;                  // 0..127
    int mat = t >> 6;                       // uniform per wave
    int tt  = t & 63;
    int tm  = (tt >> 3) << 4, tn = (tt & 7) << 4;
    const _Float16* WB = mat ? Wqv_h : Wq_h;
    const float* bias  = mat ? bqv : bq;
    _Float16* dst      = mat ? sVQ : sQ;
    v8f acc = {};
    for (int kb = 0; kb < 4; ++kb) {
      v16h a  = frag_ld(sSrc, DIM, tm, kb * 32, lane);
      v16h bb = frag_ld(WB,   DIM, tn, kb * 32, lane);
      acc = wmma_f16(a, bb, acc);
    }
    float bs = bias[tn + (lane & 15)];
#pragma unroll
    for (int j = 0; j < 8; ++j) acc[j] += bs;
    store_tile_f16(dst, DIM, tm, tn, acc, lane);
  }

  /* -- phase 2: K and V GEMMs (32x128 each, K-dim 256) -- */
  for (int i = 0; i < 4; ++i) {
    int t   = wv * 4 + i;                   // 0..31
    int mat = t >> 4;                       // uniform per wave
    int tt  = t & 15;
    int tm  = (tt >> 3) << 4, tn = (tt & 7) << 4;
    const _Float16* WB = mat ? Wv_h : Wk_h;
    const float* bias  = mat ? bv : bkk;
    _Float16* dst      = mat ? sV : sK;
    v8f acc = {};
    for (int kb = 0; kb < 8; ++kb) {
      v16h a  = frag_ld(sTgt, CH, tm, kb * 32, lane);
      v16h bb = frag_ld(WB,   CH, tn, kb * 32, lane);
      acc = wmma_f16(a, bb, acc);
    }
    float bs = bias[tn + (lane & 15)];
#pragma unroll
    for (int j = 0; j < 8; ++j) acc[j] += bs;
    store_tile_f16(dst, DIM, tm, tn, acc, lane);
  }
  __syncthreads();

  /* -- phase 3: softmax over 4-patch axis + DANE gate, per (patch, head) -- */
  {
    int p  = tid >> 3;
    int hh = tid & 7;
    int c0 = hh << 4;
    float g1[HD_], b1[HD_], wsv[HD_], cmv[HD_], kvec[HD_], vvec[HD_];
    const float* cmp = cm + (size_t)(((side * BDIM + b) * NH_) + hh) * HD_;
#pragma unroll
    for (int d = 0; d < HD_; ++d) {
      g1[d]  = ln1g[d];  b1[d] = ln1b[d];  wsv[d] = dwsw[d];
      cmv[d] = cmp[d];
    }
    ld16h(sK + p * DIM + c0, kvec);
    ld16h(sV + p * DIM + c0, vvec);

    float logits[4], smacc = 0.f;
    for (int s = 0; s < 4; ++s) {
      int row = p * 4 + s;
      float qv[HD_];
      ld16h(sQ + row * DIM + c0, qv);
      float dot = 0.f, mu = 0.f;
#pragma unroll
      for (int d = 0; d < HD_; ++d) { dot += qv[d] * kvec[d]; mu += qv[d]; }
      mu *= (1.f / HD_);
      float var = 0.f;
#pragma unroll
      for (int d = 0; d < HD_; ++d) { float dd = qv[d] - mu; var += dd * dd; }
      var *= (1.f / HD_);
      float inv = rsqrtf(var + 1e-5f);
      float sd  = 0.f;
#pragma unroll
      for (int d = 0; d < HD_; ++d)
        sd += ((qv[d] - mu) * inv * g1[d] + b1[d]) * wsv[d];
      smacc += sd;
      logits[s] = dot * 0.25f;              // d^-0.5, d=16
    }
    smacc *= 0.25f;                          // mean over 4-patch axis
    float mx = fmaxf(fmaxf(logits[0], logits[1]), fmaxf(logits[2], logits[3]));
    float e[4], ssum = 0.f;
#pragma unroll
    for (int s = 0; s < 4; ++s) { e[s] = __expf(logits[s] - mx); ssum += e[s]; }
    float rs = 1.f / ssum;
    float wg[HD_];
#pragma unroll
    for (int d = 0; d < HD_; ++d)
      wg[d] = 1.f / (1.f + __expf(-(smacc + cmv[d])));
    for (int s = 0; s < 4; ++s) {
      int row = p * 4 + s;
      float a4 = e[s] * rs * 4.f;
      float vqrow[HD_], xv[HD_];
      ld16h(sVQ + row * DIM + c0, vqrow);
#pragma unroll
      for (int d = 0; d < HD_; ++d)
        xv[d] = (1.f - wg[d]) * a4 * vvec[d] + wg[d] * vqrow[d];
      st16h(sX + row * DIM + c0, xv);
    }
  }
  __syncthreads();

  /* -- phase 4: output projection P (128x128) -- */
  for (int i = 0; i < 8; ++i) {
    int t  = wv * 8 + i;                    // 0..63
    int tm = (t >> 3) << 4, tn = (t & 7) << 4;
    v8f acc = {};
    for (int kb = 0; kb < 4; ++kb) {
      v16h a  = frag_ld(sX,   DIM, tm, kb * 32, lane);
      v16h bb = frag_ld(Wp_h, DIM, tn, kb * 32, lane);
      acc = wmma_f16(a, bb, acc);
    }
    float bs = bp[tn + (lane & 15)];
#pragma unroll
    for (int j = 0; j < 8; ++j) acc[j] += bs;
    store_tile_f32(sP, DIM, tm, tn, acc, lane);
  }
  __syncthreads();

  /* -- phase 5: LayerNorm over channels + residual + store -- */
  if (tid < 128) {
    int r = tid;
    int p = r >> 2, s = r & 3;
    int n = p0 + p, ph = n >> 7, pw = n & 127;
    int yy = ph * 2 + (s >> 1), xx = pw * 2 + (s & 1);
    float mu = 0.f;
    for (int c = 0; c < DIM; ++c) mu += sP[r * DIM + c];
    mu *= (1.f / DIM);
    float var = 0.f;
    for (int c = 0; c < DIM; ++c) { float dd = sP[r * DIM + c] - mu; var += dd * dd; }
    var *= (1.f / DIM);
    float inv = rsqrtf(var + 1e-5f);
    size_t sideoff = (size_t)side * BDIM * DIM * HH_ * WW_;
    for (int c = 0; c < DIM; ++c) {
      size_t g = (((size_t)b * DIM + c) * HH_ + yy) * WW_ + xx;
      float o  = (sP[r * DIM + c] - mu) * inv * lng[c] + lnb[c] + xh[g];
      out[sideoff + g] = o;
    }
  }
}

/* ---- launcher ----------------------------------------------------------- */

extern "C" void kernel_launch(void* const* d_in, const int* in_sizes, int n_in,
                              void* d_out, int out_size, void* d_ws, size_t ws_size,
                              hipStream_t stream) {
  (void)in_sizes; (void)n_in; (void)out_size; (void)ws_size;
  const float* x_l  = (const float*)d_in[0];
  const float* x_r  = (const float*)d_in[1];
  const float* x_lh = (const float*)d_in[2];
  const float* x_rh = (const float*)d_in[3];
  const float* Wq   = (const float*)d_in[4];
  const float* bq   = (const float*)d_in[5];
  const float* Wqv  = (const float*)d_in[6];
  const float* bqv  = (const float*)d_in[7];
  const float* Wk   = (const float*)d_in[8];
  const float* bk   = (const float*)d_in[9];
  const float* Wv   = (const float*)d_in[10];
  const float* bv   = (const float*)d_in[11];
  const float* Wp   = (const float*)d_in[12];
  const float* bp   = (const float*)d_in[13];
  const float* ln_g = (const float*)d_in[14];
  const float* ln_b = (const float*)d_in[15];
  const float* ln1g = (const float*)d_in[16];
  const float* ln1b = (const float*)d_in[17];
  const float* dwsw = (const float*)d_in[18];
  const float* dw1  = (const float*)d_in[19];
  const float* ln2g = (const float*)d_in[20];
  const float* ln2b = (const float*)d_in[21];
  const float* dw2  = (const float*)d_in[22];
  float* out = (float*)d_out;

  char* ws = (char*)d_ws;
  _Float16* wf16  = (_Float16*)ws;
  float* mean_tgt = (float*)(ws + WSB_MEAN);
  float* cm       = (float*)(ws + WSB_CM);

  k_convert<<<WSH_TOT / 256, 256, 0, stream>>>(Wq, Wqv, Wk, Wv, Wp, wf16);
  k_reduce_tgt<<<2 * BDIM, 256, 0, stream>>>(x_l, x_r, mean_tgt);
  k_cm<<<1, 64, 0, stream>>>(mean_tgt, Wk, bk, dw1, ln2g, ln2b, dw2, cm);

  int nblocks = 2 * BDIM * (N0_ / PATCHES);   // 4096
  k_pafm_main<<<nblocks, 256, SM_TOTAL, stream>>>(
      x_l, x_r, x_lh, x_rh, wf16, bq, bqv, bk, bv, bp,
      ln_g, ln_b, ln1g, ln1b, dwsw, cm, out);
}
